// RelationalConv_53489522705039
// MI455X (gfx1250) — matrix operand order repaired and checked by reading, not compiled, not run on hardware
//
#include <hip/hip_runtime.h>
#include <hip/hip_bf16.h>

// ---------------------------------------------------------------------------
// Relational GraphConv (R=4, D=128) for gfx1250 (MI455X, wave32, WMMA + TDM).
//   out[n,:] = sum_r tanh( agg[n,r,:] @ W_rel[r] + x[n,:] @ W_root[r]
//                          + b_rel[r] + b_root[r] )
//   agg[n,r,:] = sum over edges e with dst==n, attr==r of x[src(e),:]
// ---------------------------------------------------------------------------

typedef __attribute__((ext_vector_type(16))) __bf16       v16bf;
typedef __attribute__((ext_vector_type(8)))  float        v8f;
typedef __attribute__((ext_vector_type(4)))  unsigned int v4u;
typedef __attribute__((ext_vector_type(8)))  int          v8i;
typedef __attribute__((ext_vector_type(4)))  int          v4i;

#define DIM   128
#define NREL  4
// fragment table: src(2) x rel(4) x kchunk(4) x coltile(8) fragments,
// each fragment = 32 lanes x 16 bf16 halves (32 bytes/lane, contiguous)
#define NFRAG (2 * NREL * 4 * 8)
#define WAVES_PER_BLOCK 16
#define GEMM_THREADS (WAVES_PER_BLOCK * 32)

#if __has_builtin(__builtin_amdgcn_tensor_load_to_lds) && \
    __has_builtin(__builtin_amdgcn_s_wait_tensorcnt)
#define RGC_HAVE_TDM 1
#else
#define RGC_HAVE_TDM 0
#endif

__device__ __forceinline__ float rgc_tanh(float v) {
#if __has_builtin(__builtin_amdgcn_tanhf)
  return __builtin_amdgcn_tanhf(v);     // native V_TANH_F32 on gfx1250
#else
  return tanhf(v);
#endif
}

// ---------------------------- zero workspace -------------------------------
__global__ void rgc_zero_kernel(float4* __restrict__ p, size_t n4) {
  size_t i = (size_t)blockIdx.x * blockDim.x + threadIdx.x;
  size_t stride = (size_t)gridDim.x * blockDim.x;
  float4 z = make_float4(0.f, 0.f, 0.f, 0.f);
  for (; i < n4; i += stride) p[i] = z;
}

// ------------------- weight swizzle into WMMA B fragments ------------------
// B-matrix layout for v_wmma_f32_16x16x32_bf16 (32x16, K x N):
//   lane (0..15):  N = lane,    halves 0..7 = K kbase..kbase+7, 8..15 = +16..+23, kbase = 32c
//   lane (16..31): N = lane-16, same with kbase = 32c+8
__global__ void rgc_prep_kernel(const float* __restrict__ W_rel,
                                const float* __restrict__ W_root,
                                const float* __restrict__ b_rel,
                                const float* __restrict__ b_root,
                                __bf16* __restrict__ bfrag,
                                float* __restrict__ bsum) {
  int t = blockIdx.x * blockDim.x + threadIdx.x;
  if (t < NFRAG * 32) {
    int lane = t & 31;
    int frag = t >> 5;                 // (((src*4 + r)*4 + c)*8 + ct)
    int ct  = frag & 7;
    int c   = (frag >> 3) & 3;
    int r   = (frag >> 5) & 3;
    int src = (frag >> 7) & 1;
    const float* W = src ? W_root : W_rel;
    int n     = ct * 16 + (lane & 15);
    int kbase = 32 * c + ((lane >> 4) & 1) * 8;
    __bf16* dst = bfrag + (size_t)t * 16;
#pragma unroll
    for (int j = 0; j < 16; ++j)
      dst[j] = (__bf16)W[((size_t)r * DIM + (kbase + j + (j >= 8 ? 8 : 0))) * DIM + n];
  } else {
    int i = t - NFRAG * 32;
    if (i < NREL * DIM) bsum[i] = b_rel[i] + b_root[i];
  }
}

// ------------------------------ edge scatter -------------------------------
// one wave32 per edge: gather x[src] (float4/lane), atomic-add into agg[dst][r]
__global__ void rgc_scatter_kernel(const float* __restrict__ x,
                                   const int* __restrict__ edge_index,
                                   const int* __restrict__ edge_attr,
                                   float* __restrict__ agg, int E) {
  int gid  = blockIdx.x * blockDim.x + threadIdx.x;
  int e    = gid >> 5;
  if (e >= E) return;
  int lane = gid & 31;
  int src  = edge_index[e];
  int dst  = edge_index[E + e];
  int r    = edge_attr[e];
  float4 v = ((const float4*)(x + (size_t)src * DIM))[lane];
  float* o = agg + (((size_t)dst * NREL + r) * DIM) + lane * 4;
  unsafeAtomicAdd(o + 0, v.x);
  unsafeAtomicAdd(o + 1, v.y);
  unsafeAtomicAdd(o + 2, v.z);
  unsafeAtomicAdd(o + 3, v.w);
}

// ----------------------------- WMMA GEMM pass ------------------------------
// A-matrix layout for v_wmma_f32_16x16x32_bf16 (16x32, M x K):
//   lane (0..15):  M = lane, halves 0..7 = K kbase..kbase+7, 8..15 = +16..+23,
//                  kbase = 32c       (lanes 16..31: kbase = 32c + 8)
__device__ __forceinline__ v16bf rgc_load_a(const float* __restrict__ p) {
  float4 f0 = ((const float4*)p)[0];
  float4 f1 = ((const float4*)p)[1];
  float4 f2 = ((const float4*)(p + 16))[0];
  float4 f3 = ((const float4*)(p + 16))[1];
  v16bf a;
  a[0]  = (__bf16)f0.x; a[1]  = (__bf16)f0.y; a[2]  = (__bf16)f0.z; a[3]  = (__bf16)f0.w;
  a[4]  = (__bf16)f1.x; a[5]  = (__bf16)f1.y; a[6]  = (__bf16)f1.z; a[7]  = (__bf16)f1.w;
  a[8]  = (__bf16)f2.x; a[9]  = (__bf16)f2.y; a[10] = (__bf16)f2.z; a[11] = (__bf16)f2.w;
  a[12] = (__bf16)f3.x; a[13] = (__bf16)f3.y; a[14] = (__bf16)f3.z; a[15] = (__bf16)f3.w;
  return a;
}

__global__ void __launch_bounds__(GEMM_THREADS)
rgc_gemm_kernel(const float* __restrict__ x,
                const float* __restrict__ agg,
                const __bf16* __restrict__ bfrag,
                const float* __restrict__ bsum,
                float* __restrict__ out,
                int nTiles, int Nnodes) {
  // per-relation B-fragment slice staged in LDS: 64 frags x 1 KB = 64 KB
  __shared__ __bf16 sB[64 * 32 * 16];

  int waveInBlk = (int)(threadIdx.x >> 5);
  int lane      = (int)(threadIdx.x & 31);
  int tile      = blockIdx.x * WAVES_PER_BLOCK + waveInBlk;
  bool active   = tile < nTiles;
  int tclamp    = active ? tile : (nTiles - 1);   // inactive waves shadow the
  int node0     = tclamp * 16;                    // last tile (barriers only)
  int m  = lane & 15;
  int hi = (lane >> 4) & 1;
  int row = node0 + m;
  if (row >= Nnodes) row = Nnodes - 1;
  const float* xrow = x + (size_t)row * DIM;

  // hoist root-input fragments: x row converted to bf16 once, reused per r
  v16bf aX[4];
#pragma unroll
  for (int c = 0; c < 4; ++c) aX[c] = rgc_load_a(xrow + 32 * c + hi * 8);

  v8f outsum[8];
#pragma unroll
  for (int ct = 0; ct < 8; ++ct) outsum[ct] = (v8f)(0.f);

  for (int r = 0; r < NREL; ++r) {
    __syncthreads();   // previous iteration's LDS reads complete
#if RGC_HAVE_TDM
    // Tensor Data Mover: one wave issues a 2-row DMA (src0 slice + src1 slice,
    // 32 KB each, row stride 128 KB) straight into LDS. D# per ISA 08 §8.
    if (waveInBlk == 0) {
      unsigned lds_addr = (unsigned)(uintptr_t)(void*)sB;
      unsigned long long ga =
          (unsigned long long)(uintptr_t)bfrag + (size_t)r * 32 * 1024;
      v4u g0;
      g0[0] = 1u;                                  // count=1, user desc
      g0[1] = lds_addr;                            // lds_addr[31:0]
      g0[2] = (unsigned)(ga & 0xffffffffu);        // global_addr[31:0]
      g0[3] = (unsigned)((ga >> 32) & 0x01ffffffu) // global_addr[56:32]
              | (2u << 30);                        // type=2 ("image")
      v8i g1;
      g1[0] = 0x00020000;          // workgroup_mask=0, data_size=2 (4 bytes)
      g1[1] = (int)(8192u << 16);  // tensor_dim0[15:0]=8192 (32KB of dwords)
      g1[2] = (int)(2u << 16);     // tensor_dim0[31:16]=0, tensor_dim1=2
      g1[3] = (int)(8192u << 16);  // tile_dim0=8192
      g1[4] = 2;                   // tile_dim1=2, tile_dim2=0
      g1[5] = 32768;               // tensor_dim0_stride = 128KB in dwords
      g1[6] = 0;
      g1[7] = 0;
      v4i g2 = {0, 0, 0, 0};       // unused for 2-D tensor
      v4i g3 = {0, 0, 0, 0};
#if __clang_major__ >= 23
      v8i g4 = {0, 0, 0, 0, 0, 0, 0, 0};
      __builtin_amdgcn_tensor_load_to_lds(g0, g1, g2, g3, g4, 0);
#else
      __builtin_amdgcn_tensor_load_to_lds(g0, g1, g2, g3, 0);
#endif
      __builtin_amdgcn_s_wait_tensorcnt(0);
    }
#else
    {
      // fallback: cooperative copy of src0 frags [r*32,+32) and src1 [128+r*32,+32)
      const uint4* g0 = (const uint4*)bfrag + (size_t)(r * 32) * 64;
      const uint4* g1 = (const uint4*)bfrag + (size_t)(128 + r * 32) * 64;
      uint4* s = (uint4*)sB;
      for (int i = threadIdx.x; i < 2048; i += GEMM_THREADS) {
        s[i]        = g0[i];
        s[i + 2048] = g1[i];
      }
    }
#endif
    __syncthreads();

    const float* arow = agg + ((size_t)row * NREL + r) * DIM;
    v8f acc[8];
#pragma unroll
    for (int ct = 0; ct < 8; ++ct) acc[ct] = (v8f)(0.f);

#pragma unroll
    for (int c = 0; c < 4; ++c) {
      v16bf aAgg = rgc_load_a(arow + 32 * c + hi * 8);
      const __bf16* sp = sB + ((size_t)(c * 8) * 32 + lane) * 16;        // rel
      const __bf16* sq = sB + ((size_t)(32 + c * 8) * 32 + lane) * 16;   // root
#pragma unroll
      for (int ct = 0; ct < 8; ++ct) {
        v16bf bRel  = *(const v16bf*)(sp + (size_t)ct * 32 * 16);
        v16bf bRoot = *(const v16bf*)(sq + (size_t)ct * 32 * 16);
        acc[ct] = __builtin_amdgcn_wmma_f32_16x16x32_bf16(
            false, aAgg, false, bRel, (short)0, acc[ct], false, false);
        acc[ct] = __builtin_amdgcn_wmma_f32_16x16x32_bf16(
            false, aX[c], false, bRoot, (short)0, acc[ct], false, false);
      }
    }
    // bias + tanh + accumulate over relations (C/D layout: lane->N, vgpr i->M)
#pragma unroll
    for (int ct = 0; ct < 8; ++ct) {
      float bias = bsum[r * DIM + ct * 16 + m];
      v8f a = acc[ct];
      v8f o = outsum[ct];
#pragma unroll
      for (int i = 0; i < 8; ++i) o[i] += rgc_tanh(a[i] + bias);
      outsum[ct] = o;
    }
  }

  if (active) {
    // store: D layout lane 0-15 -> N=lane, vgpr i -> M=i ; lanes 16-31 -> M=i+8
#pragma unroll
    for (int i = 0; i < 8; ++i) {
      int rowM = node0 + i + hi * 8;
      if (rowM >= Nnodes) continue;
      float* orow = out + (size_t)rowM * DIM + m;
#pragma unroll
      for (int ct = 0; ct < 8; ++ct) orow[ct * 16] = outsum[ct][i];
    }
  }
}

// ------------------------------- launcher ----------------------------------
extern "C" void kernel_launch(void* const* d_in, const int* in_sizes, int n_in,
                              void* d_out, int out_size, void* d_ws, size_t ws_size,
                              hipStream_t stream) {
  const float* x      = (const float*)d_in[0];
  const int*   ei     = (const int*)d_in[1];
  const int*   ea     = (const int*)d_in[2];
  const float* W_rel  = (const float*)d_in[3];
  const float* b_rel  = (const float*)d_in[4];
  const float* W_root = (const float*)d_in[5];
  const float* b_root = (const float*)d_in[6];
  float* out = (float*)d_out;

  const int Nn = in_sizes[0] / DIM;     // 100000
  const int E  = in_sizes[2];           // 640000

  // workspace layout: [agg f32 N*R*D] [bfrag bf16 NFRAG*32*16] [bsum f32 R*D]
  size_t aggElems = (size_t)Nn * NREL * DIM;
  float* agg = (float*)d_ws;
  size_t off = (aggElems * sizeof(float) + 255) & ~(size_t)255;
  __bf16* bfrag = (__bf16*)((char*)d_ws + off);
  float*  bsum  = (float*)((char*)d_ws + off + (size_t)NFRAG * 32 * 16 * sizeof(__bf16));

  // 1) zero the aggregation buffer
  rgc_zero_kernel<<<4096, 256, 0, stream>>>((float4*)agg, aggElems / 4);

  // 2) swizzle weights into bf16 WMMA B fragments + fused bias
  int prepThreads = NFRAG * 32 + NREL * DIM;
  rgc_prep_kernel<<<(prepThreads + 255) / 256, 256, 0, stream>>>(
      W_rel, W_root, b_rel, b_root, bfrag, bsum);

  // 3) edge gather + segment-sum scatter (one wave per edge)
  long long sthreads = (long long)E * 32;
  rgc_scatter_kernel<<<(unsigned)((sthreads + 255) / 256), 256, 0, stream>>>(
      x, ei, ea, agg, E);

  // 4) WMMA GEMM + tanh + relation reduction; 16 tiles per block, B via TDM->LDS
  int nTiles = (Nn + 15) / 16;
  int blocks = (nTiles + WAVES_PER_BLOCK - 1) / WAVES_PER_BLOCK;
  rgc_gemm_kernel<<<blocks, GEMM_THREADS, 0, stream>>>(
      x, agg, bfrag, bsum, out, nTiles, Nn);
}
